// Qwen2MoELayer_38757784879530
// MI455X (gfx1250) — compile-verified, matrix operand order checked by reading, not run on hardware
//
#include <hip/hip_runtime.h>
#include <hip/hip_bf16.h>
#include <math.h>

// Problem constants (match reference)
#define T_TOK 8192
#define HID   2048
#define NEXP  32
#define TOPK  4
#define IDIM  1024
#define TWOI  (2*IDIM)
#define NSLOT (T_TOK*TOPK)          // 32768 dispatched (token,expert) slots
#define TM    128                   // row tile (slots)
#define TKS   32                    // bf16 WMMA K depth
#define ASTR  40                    // LDS row stride in bf16 (32 data + 8 pad) = 80B = 5*uint4
#define MAXTILES (NSLOT/TM + NEXP)  // 288 worst case

typedef __bf16 bf16;
typedef __attribute__((ext_vector_type(16))) __bf16 v16bf;
typedef __attribute__((ext_vector_type(8)))  float  v8f;
typedef __attribute__((ext_vector_type(4)))  int    v4i;

#define AS1 __attribute__((address_space(1)))
#define AS3 __attribute__((address_space(3)))

#if defined(__has_builtin)
#if __has_builtin(__builtin_amdgcn_global_load_async_to_lds_b128) && \
    __has_builtin(__builtin_amdgcn_s_wait_asynccnt)
#define HAVE_ASYNC_LDS 1
#endif
#endif
#ifndef HAVE_ASYNC_LDS
#define HAVE_ASYNC_LDS 0
#endif

union FragAB { uint4 q[2]; v16bf v; };

struct WS {
  int   r_idx[NSLOT];
  float r_w[NSLOT];
  int   counts[NEXP];
  int   cursor[NEXP];
  int   offsets[NEXP];
  int   ntiles;
  int   pad0;
  int   tile_e[MAXTILES];
  int   tile_start[MAXTILES];
  int   tile_valid[MAXTILES];
  int   slot_token[NSLOT];
  float slot_w[NSLOT];
};

// ---------------- kernel 1: zero output + counts ----------------
__global__ void zero_kernel(float* __restrict__ out, int n, int* __restrict__ counts) {
  for (int i = blockIdx.x * blockDim.x + threadIdx.x; i < n; i += gridDim.x * blockDim.x)
    out[i] = 0.f;
  if (blockIdx.x == 0 && threadIdx.x < NEXP) counts[threadIdx.x] = 0;
}

// ---------------- kernel 2: router (logits -> softmax -> top4) ----------------
// Block = 8 tokens (8 waves). Router weights + token rows staged through LDS so
// global reads are coalesced; per-lane dot uses LDS broadcast for the token row.
__global__ __launch_bounds__(256)
void router_topk_kernel(const float* __restrict__ hs, const float* __restrict__ rw,
                        int* __restrict__ r_idx, float* __restrict__ r_w) {
  __shared__ float rwbuf[NEXP][133];    // pad 133 (odd) -> conflict-free column reads
  __shared__ float hbuf[8][128];
  const int tid = threadIdx.x, lane = tid & 31, wave = tid >> 5;
  const int token = blockIdx.x * 8 + wave;
  float acc = 0.f;
  for (int k0 = 0; k0 < HID; k0 += 128) {
    __syncthreads();
    for (int i = tid; i < NEXP * 128; i += 256) {
      int e = i >> 7, k = i & 127;
      rwbuf[e][k] = rw[(size_t)e * HID + k0 + k];
    }
    for (int i = tid; i < 8 * 128; i += 256) {
      int t = i >> 7, k = i & 127;
      hbuf[t][k] = hs[(size_t)(blockIdx.x * 8 + t) * HID + k0 + k];
    }
    __syncthreads();
#pragma unroll 8
    for (int k = 0; k < 128; ++k) acc += hbuf[wave][k] * rwbuf[lane][k];
  }
  // wave32 softmax over experts (lane == expert)
  float m = acc;
  for (int off = 16; off; off >>= 1) m = fmaxf(m, __shfl_xor(m, off, 32));
  float p = __expf(acc - m);
  float s = p;
  for (int off = 16; off; off >>= 1) s += __shfl_xor(s, off, 32);
  p /= s;
  // top-4 via repeated wave argmax
  for (int k = 0; k < TOPK; ++k) {
    float v = p; int idx = lane;
    for (int off = 16; off; off >>= 1) {
      float ov = __shfl_xor(v, off, 32);
      int   oi = __shfl_xor(idx, off, 32);
      if (ov > v || (ov == v && oi < idx)) { v = ov; idx = oi; }
    }
    if (lane == 0) { r_idx[token*TOPK + k] = idx; r_w[token*TOPK + k] = v; }
    if (lane == idx) p = -1.f;
  }
}

// ---------------- kernel 3: per-expert histogram ----------------
__global__ void hist_kernel(const int* __restrict__ r_idx, int* __restrict__ counts) {
  int i = blockIdx.x * 256 + threadIdx.x;
  if (i < NSLOT) atomicAdd(&counts[r_idx[i]], 1);
}

// ---------------- kernel 4: scan + build row-tile descriptors ----------------
__global__ void scan_desc_kernel(const int* __restrict__ counts, int* __restrict__ offsets,
                                 int* __restrict__ cursor, int* __restrict__ ntiles,
                                 int* __restrict__ te, int* __restrict__ ts, int* __restrict__ tv) {
  if (threadIdx.x == 0) {
    int off = 0, nt = 0;
    for (int e = 0; e < NEXP; ++e) {
      offsets[e] = off; cursor[e] = off;
      int c = counts[e];
      for (int r = 0; r < c; r += TM) {
        te[nt] = e; ts[nt] = off + r;
        int rem = c - r; tv[nt] = rem < TM ? rem : TM;
        ++nt;
      }
      off += c;
    }
    *ntiles = nt;
  }
}

// ---------------- kernel 5: slot assignment (dispatch permutation) ----------------
__global__ void assign_kernel(const int* __restrict__ r_idx, const float* __restrict__ r_w,
                              int* __restrict__ cursor, int* __restrict__ slot_token,
                              float* __restrict__ slot_w) {
  int i = blockIdx.x * 256 + threadIdx.x;
  if (i < NSLOT) {
    int e = r_idx[i];
    int slot = atomicAdd(&cursor[e], 1);
    slot_token[slot] = i >> 2;        // i / TOPK
    slot_w[slot]     = r_w[i];
  }
}

// ---------------- kernel 6: grouped GEMM gate/up + SiLU-GLU, bf16 WMMA ----------------
__global__ __launch_bounds__(256)
void gemm_gateup_silu_kernel(const float* __restrict__ hidden, const float* __restrict__ gup,
                             const int* __restrict__ ntiles, const int* __restrict__ te,
                             const int* __restrict__ ts, const int* __restrict__ tv,
                             const int* __restrict__ slot_token, bf16* __restrict__ act) {
  const int tile = blockIdx.x;
  if (tile >= *ntiles) return;
  const int e = te[tile], start = ts[tile], valid = tv[tile];
  const int n0 = blockIdx.y * 128;              // column base within IDIM

  __shared__ alignas(16) bf16 As[TM * ASTR];    // A tile  [M][K], row-major
  __shared__ alignas(16) bf16 Bg[128 * ASTR];   // gate B tile, transposed [N][K]
  __shared__ alignas(16) bf16 Bu[128 * ASTR];   // up   B tile, transposed [N][K]
  __shared__ int s_tok[TM];

  const int tid  = threadIdx.x;
  const int lane = tid & 31, wave = tid >> 5;
  const int wm = wave >> 1, wn = wave & 1;      // 4x2 wave grid; wave tile 32x64
  const int ml = lane & 15, g = lane >> 4;

  if (tid < TM) s_tok[tid] = (tid < valid) ? slot_token[start + tid] : 0;
  __syncthreads();

  // Hoist per-thread staging pointers OUT of the K loop (tokens are K-invariant).
  const float* arow[4];
  bf16*        adst[4];
#pragma unroll
  for (int i = 0; i < 4; ++i) {
    int c = tid + 256 * i;                       // 1024 float4 chunks of the 128x32 A tile
    int row = c >> 3, kq = c & 7;
    arow[i] = hidden + (size_t)s_tok[row] * HID + kq * 4;
    adst[i] = As + row * ASTR + kq * 4;
  }
  const int   bn = tid & 127;                    // B staging: column within tile
  const int   bk = (tid >> 7) * 16;              // first of 16 K rows handled
  const float* bbase = gup + (size_t)e * HID * TWOI + n0 + bn;

  v8f accg[2][4], accu[2][4];
  for (int i = 0; i < 2; ++i)
    for (int j = 0; j < 4; ++j)
      for (int c = 0; c < 8; ++c) { accg[i][j][c] = 0.f; accu[i][j][c] = 0.f; }

  const uint4* As4 = reinterpret_cast<const uint4*>(As);
  const uint4* Bg4 = reinterpret_cast<const uint4*>(Bg);
  const uint4* Bu4 = reinterpret_cast<const uint4*>(Bu);

  for (int k0 = 0; k0 < HID; k0 += TKS) {
    // stage A: issue all 4 global float4 loads, then convert+store (bf16 pairs)
    float4 av[4];
#pragma unroll
    for (int i = 0; i < 4; ++i) av[i] = *reinterpret_cast<const float4*>(arow[i] + k0);
#pragma unroll
    for (int i = 0; i < 4; ++i) {
      bf16* d = adst[i];
      d[0] = (bf16)av[i].x; d[1] = (bf16)av[i].y; d[2] = (bf16)av[i].z; d[3] = (bf16)av[i].w;
    }
    // stage B gate+up: 16 coalesced rows each, loads batched before LDS stores
    float bgv[16], buv[16];
#pragma unroll
    for (int i = 0; i < 16; ++i) {
      const float* src = bbase + (size_t)(k0 + bk + i) * TWOI;
      bgv[i] = src[0];
      buv[i] = src[IDIM];
    }
#pragma unroll
    for (int i = 0; i < 16; ++i) {
      Bg[bn * ASTR + bk + i] = (bf16)bgv[i];
      Bu[bn * ASTR + bk + i] = (bf16)buv[i];
    }
    if (k0 + TKS < HID)   // hint next K-step weight rows -> global_prefetch_b8
      __builtin_prefetch(bbase + (size_t)(k0 + TKS + bk) * TWOI, 0, 1);
    __syncthreads();

    // fragments: lanes 0-15 -> K0-7 & K16-23, lanes 16-31 -> K8-15 & K24-31 (ISA layout)
    FragAB a[2], bg[4], bu[4];
#pragma unroll
    for (int sm = 0; sm < 2; ++sm) {
      int r5 = (wm * 32 + sm * 16 + ml) * 5;
      a[sm].q[0] = As4[r5 + g]; a[sm].q[1] = As4[r5 + 2 + g];
    }
#pragma unroll
    for (int sn = 0; sn < 4; ++sn) {
      int r5 = (wn * 64 + sn * 16 + ml) * 5;
      bg[sn].q[0] = Bg4[r5 + g]; bg[sn].q[1] = Bg4[r5 + 2 + g];
      bu[sn].q[0] = Bu4[r5 + g]; bu[sn].q[1] = Bu4[r5 + 2 + g];
    }
#pragma unroll
    for (int sm = 0; sm < 2; ++sm)
#pragma unroll
      for (int sn = 0; sn < 4; ++sn) {
        accg[sm][sn] = __builtin_amdgcn_wmma_f32_16x16x32_bf16(
            false, a[sm].v, false, bg[sn].v, (short)0, accg[sm][sn], false, false);
        accu[sm][sn] = __builtin_amdgcn_wmma_f32_16x16x32_bf16(
            false, a[sm].v, false, bu[sn].v, (short)0, accu[sm][sn], false, false);
      }
    __syncthreads();
  }

  // epilogue: act = silu(gate) * up, store bf16 (C/D layout: N=lane%16, M=v+8*(lane/16))
  for (int sm = 0; sm < 2; ++sm)
    for (int sn = 0; sn < 4; ++sn)
      for (int v = 0; v < 8; ++v) {
        int mloc = wm * 32 + sm * 16 + v + 8 * g;
        if (mloc < valid) {
          float gv = accg[sm][sn][v], uv = accu[sm][sn][v];
          float avv = gv / (1.f + __expf(-gv)) * uv;
          act[(size_t)(start + mloc) * IDIM + n0 + wn * 64 + sn * 16 + ml] = (bf16)avv;
        }
      }
}

// ---------------- kernel 7: grouped GEMM down + weighted scatter-add ----------------
__global__ __launch_bounds__(256)
void gemm_down_scatter_kernel(const bf16* __restrict__ act, const float* __restrict__ dwn,
                              const int* __restrict__ ntiles, const int* __restrict__ te,
                              const int* __restrict__ ts, const int* __restrict__ tv,
                              const int* __restrict__ slot_token, const float* __restrict__ slot_w,
                              float* __restrict__ out) {
  const int tile = blockIdx.x;
  if (tile >= *ntiles) return;
  const int e = te[tile], start = ts[tile], valid = tv[tile];
  const int n0 = blockIdx.y * 128;              // column base within HID

  __shared__ alignas(16) bf16 As[TM * ASTR];
  __shared__ alignas(16) bf16 Bt[128 * ASTR];
  __shared__ int   s_tok[TM];
  __shared__ float s_w[TM];

  const int tid  = threadIdx.x;
  const int lane = tid & 31, wave = tid >> 5;
  const int wm = wave >> 1, wn = wave & 1;
  const int ml = lane & 15, g = lane >> 4;

  if (tid < TM) {
    s_tok[tid] = (tid < valid) ? slot_token[start + tid] : 0;
    s_w[tid]   = (tid < valid) ? slot_w[start + tid] : 0.f;
  }
  __syncthreads();

  // Hoisted per-thread staging pointers (2 x 16B chunks of the 128x32 bf16 A tile)
  const bf16* asrc[2];
  bf16*       adst[2];
#pragma unroll
  for (int i = 0; i < 2; ++i) {
    int c = tid + 256 * i;
    int row = c >> 2, kq = c & 3;
    int r2 = (row < valid) ? (start + row) : start;   // padded rows: harmless re-read
    asrc[i] = act + (size_t)r2 * IDIM + kq * 8;
    adst[i] = As + row * ASTR + kq * 8;
  }
  const int   bn = tid & 127;
  const int   bk = (tid >> 7) * 16;
  const float* bbase = dwn + (size_t)e * IDIM * HID + n0 + bn;

  v8f acc[2][4];
  for (int i = 0; i < 2; ++i)
    for (int j = 0; j < 4; ++j)
      for (int c = 0; c < 8; ++c) acc[i][j][c] = 0.f;

  const uint4* As4 = reinterpret_cast<const uint4*>(As);
  const uint4* Bt4 = reinterpret_cast<const uint4*>(Bt);

  for (int k0 = 0; k0 < IDIM; k0 += TKS) {
    // stage A: act is already bf16 -> pure LDS<-global copy. Use CDNA5 async
    // load-to-LDS (ASYNCcnt path): builtin takes typed (global v4i*, lds v4i*).
#if HAVE_ASYNC_LDS
#pragma unroll
    for (int i = 0; i < 2; ++i)
      __builtin_amdgcn_global_load_async_to_lds_b128(
          (AS1 v4i*)(asrc[i] + k0), (AS3 v4i*)adst[i], 0, 0);
#else
#pragma unroll
    for (int i = 0; i < 2; ++i) {
      uint4 avv = *reinterpret_cast<const uint4*>(asrc[i] + k0);
      *reinterpret_cast<uint4*>(adst[i]) = avv;
    }
#endif
    // stage B: fp32 -> bf16, transposed [N][K]; loads batched before LDS stores
    float bv[16];
#pragma unroll
    for (int i = 0; i < 16; ++i) bv[i] = bbase[(size_t)(k0 + bk + i) * HID];
#pragma unroll
    for (int i = 0; i < 16; ++i) Bt[bn * ASTR + bk + i] = (bf16)bv[i];
    if (k0 + TKS < IDIM)
      __builtin_prefetch(bbase + (size_t)(k0 + TKS + bk) * HID, 0, 1);
#if HAVE_ASYNC_LDS
    __builtin_amdgcn_s_wait_asynccnt(0);
#endif
    __syncthreads();

    FragAB a[2], b[4];
#pragma unroll
    for (int sm = 0; sm < 2; ++sm) {
      int r5 = (wm * 32 + sm * 16 + ml) * 5;
      a[sm].q[0] = As4[r5 + g]; a[sm].q[1] = As4[r5 + 2 + g];
    }
#pragma unroll
    for (int sn = 0; sn < 4; ++sn) {
      int r5 = (wn * 64 + sn * 16 + ml) * 5;
      b[sn].q[0] = Bt4[r5 + g]; b[sn].q[1] = Bt4[r5 + 2 + g];
    }
#pragma unroll
    for (int sm = 0; sm < 2; ++sm)
#pragma unroll
      for (int sn = 0; sn < 4; ++sn)
        acc[sm][sn] = __builtin_amdgcn_wmma_f32_16x16x32_bf16(
            false, a[sm].v, false, b[sn].v, (short)0, acc[sm][sn], false, false);
    __syncthreads();
  }

  // epilogue: weighted scatter-add back to token order
  for (int sm = 0; sm < 2; ++sm)
    for (int sn = 0; sn < 4; ++sn)
      for (int v = 0; v < 8; ++v) {
        int mloc = wm * 32 + sm * 16 + v + 8 * g;
        if (mloc < valid) {
          float val = acc[sm][sn][v] * s_w[mloc];
          atomicAdd(&out[(size_t)s_tok[mloc] * HID + n0 + wn * 64 + sn * 16 + ml], val);
        }
      }
}

// ---------------- launch ----------------
extern "C" void kernel_launch(void* const* d_in, const int* in_sizes, int n_in,
                              void* d_out, int out_size, void* d_ws, size_t ws_size,
                              hipStream_t stream) {
  const float* hs  = (const float*)d_in[0];   // [T, H]
  const float* rw  = (const float*)d_in[1];   // [E, H]
  const float* gup = (const float*)d_in[2];   // [E, H, 2I]
  const float* dwn = (const float*)d_in[3];   // [E, I, H]
  float* out = (float*)d_out;                 // [T, H]

  char* wsb = (char*)d_ws;
  WS* W = (WS*)wsb;
  bf16* act = (bf16*)(wsb + ((sizeof(WS) + 255) & ~(size_t)255));   // [NSLOT, IDIM] bf16, 64 MB

  zero_kernel<<<dim3(1024), 256, 0, stream>>>(out, T_TOK * HID, W->counts);
  router_topk_kernel<<<dim3(T_TOK / 8), 256, 0, stream>>>(hs, rw, W->r_idx, W->r_w);
  hist_kernel<<<dim3(NSLOT / 256), 256, 0, stream>>>(W->r_idx, W->counts);
  scan_desc_kernel<<<1, 32, 0, stream>>>(W->counts, W->offsets, W->cursor, &W->ntiles,
                                         W->tile_e, W->tile_start, W->tile_valid);
  assign_kernel<<<dim3(NSLOT / 256), 256, 0, stream>>>(W->r_idx, W->r_w, W->cursor,
                                                       W->slot_token, W->slot_w);
  gemm_gateup_silu_kernel<<<dim3(MAXTILES, IDIM / 128), 256, 0, stream>>>(
      hs, gup, &W->ntiles, W->tile_e, W->tile_start, W->tile_valid, W->slot_token, act);
  gemm_down_scatter_kernel<<<dim3(MAXTILES, HID / 128), 256, 0, stream>>>(
      act, dwn, &W->ntiles, W->tile_e, W->tile_start, W->tile_valid,
      W->slot_token, W->slot_w, out);
}
